// CrossAttBlock_19396072308837
// MI455X (gfx1250) — matrix operand order repaired and checked by reading, not compiled, not run on hardware
//
#include <hip/hip_runtime.h>
#include <hip/hip_bf16.h>

// ---------- types ----------
typedef __attribute__((ext_vector_type(16))) __bf16 v16bf;
typedef __attribute__((ext_vector_type(8)))  float  v8f;
typedef __attribute__((ext_vector_type(4)))  float  f4_t;
typedef __attribute__((ext_vector_type(4)))  unsigned int u4_t;
typedef __attribute__((ext_vector_type(2)))  unsigned int u2_t;

union AFrag { v16bf v; f4_t f[2]; };

#define NB 32
#define NLP 2048   // large patch tokens per batch
#define NSP 512    // small patch tokens per batch
#define DLRG 512   // large-branch token dim
#define DSML 1024  // small-branch token dim

__device__ __forceinline__ unsigned short f2bf(float f) {
  unsigned u = __float_as_uint(f);
  unsigned r = u + 0x7FFFu + ((u >> 16) & 1u);
  return (unsigned short)(r >> 16);
}
__device__ __forceinline__ float bf2f(unsigned short h) {
  return __uint_as_float(((unsigned)h) << 16);
}
__device__ __forceinline__ float wave_sum(float v) {
#pragma unroll
  for (int o = 16; o; o >>= 1) v += __shfl_xor(v, o, 32);
  return v;
}

// ---------- CDNA5 async global->LDS (ASYNCcnt-tracked, no VGPR data path) ----------
// Flat LDS pointers carry the LDS byte offset in addr[31:0] (ISA 10.2 aperture map).
__device__ __forceinline__ void async_g2l_b128(void* lds_ptr, const void* gptr) {
  unsigned off = (unsigned)(unsigned long long)lds_ptr;
  asm volatile("global_load_async_to_lds_b128 %0, %1, off"
               :: "v"(off), "v"(gptr) : "memory");
}
__device__ __forceinline__ void wait_async8() {
  asm volatile("s_wait_asynccnt 0x8" ::: "memory");
}
__device__ __forceinline__ void wait_async0() {
  asm volatile("s_wait_asynccnt 0x0" ::: "memory");
}

// ---------- prep: large patch tokens: copy out, bf16 raw, bf16 normalized ----------
__global__ void prep_large(const float* __restrict__ xl, float* __restrict__ out0,
                           unsigned short* __restrict__ xlb, unsigned short* __restrict__ xhl) {
  int tok = blockIdx.x;              // 0..NB*NLP-1
  int b = tok >> 11, j = tok & 2047;
  const f4_t* src = (const f4_t*)(xl + ((size_t)b * 2049 + 1 + j) * DLRG);
  f4_t v = src[threadIdx.x];         // 128 threads * 4 = 512
  ((f4_t*)out0)[(size_t)tok * 128 + threadIdx.x] = v;
  __shared__ float s1[128], s2[128], mr[2];
  float sm = v.x + v.y + v.z + v.w;
  float sq = v.x * v.x + v.y * v.y + v.z * v.z + v.w * v.w;
  int t = threadIdx.x;
  s1[t] = sm; s2[t] = sq; __syncthreads();
  for (int s = 64; s; s >>= 1) {
    if (t < s) { s1[t] += s1[t + s]; s2[t] += s2[t + s]; }
    __syncthreads();
  }
  if (!t) {
    float m = s1[0] * (1.f / 512.f);
    float var = s2[0] * (1.f / 512.f) - m * m;
    mr[0] = m; mr[1] = rsqrtf(var + 1e-5f);
  }
  __syncthreads();
  float m = mr[0], rs = mr[1];
  u2_t raw, nh;
  raw.x = (unsigned)f2bf(v.x) | ((unsigned)f2bf(v.y) << 16);
  raw.y = (unsigned)f2bf(v.z) | ((unsigned)f2bf(v.w) << 16);
  nh.x = (unsigned)f2bf((v.x - m) * rs) | ((unsigned)f2bf((v.y - m) * rs) << 16);
  nh.y = (unsigned)f2bf((v.z - m) * rs) | ((unsigned)f2bf((v.w - m) * rs) << 16);
  ((u2_t*)xlb)[(size_t)tok * 128 + t] = raw;
  ((u2_t*)xhl)[(size_t)tok * 128 + t] = nh;
}

// ---------- prep: small patch tokens: copy out, bf16 normalized ----------
__global__ void prep_small(const float* __restrict__ xs, float* __restrict__ out1,
                           unsigned short* __restrict__ xhs) {
  int tok = blockIdx.x;              // 0..NB*NSP-1
  int b = tok >> 9, j = tok & 511;
  const f4_t* src = (const f4_t*)(xs + ((size_t)b * 513 + 1 + j) * DSML);
  f4_t v = src[threadIdx.x];         // 256 threads * 4 = 1024
  ((f4_t*)out1)[(size_t)tok * 256 + threadIdx.x] = v;
  __shared__ float s1[256], s2[256], mr[2];
  float sm = v.x + v.y + v.z + v.w;
  float sq = v.x * v.x + v.y * v.y + v.z * v.z + v.w * v.w;
  int t = threadIdx.x;
  s1[t] = sm; s2[t] = sq; __syncthreads();
  for (int s = 128; s; s >>= 1) {
    if (t < s) { s1[t] += s1[t + s]; s2[t] += s2[t + s]; }
    __syncthreads();
  }
  if (!t) {
    float m = s1[0] * (1.f / 1024.f);
    float var = s2[0] * (1.f / 1024.f) - m * m;
    mr[0] = m; mr[1] = rsqrtf(var + 1e-5f);
  }
  __syncthreads();
  float m = mr[0], rs = mr[1];
  u2_t nh;
  nh.x = (unsigned)f2bf((v.x - m) * rs) | ((unsigned)f2bf((v.y - m) * rs) << 16);
  nh.y = (unsigned)f2bf((v.z - m) * rs) | ((unsigned)f2bf((v.w - m) * rs) << 16);
  ((u2_t*)xhs)[(size_t)tok * 256 + t] = nh;
}

// ---------- column max over small patch tokens ----------
__global__ void smax_kernel(const float* __restrict__ xs, float* __restrict__ smaxp) {
  int c = blockIdx.x * 256 + threadIdx.x;   // 0..1023
  int b = blockIdx.y;
  float m = -3.402823466e38f;
  const float* base = xs + ((size_t)b * 513 + 1) * DSML + c;
  for (int j = 0; j < NSP; ++j) m = fmaxf(m, base[(size_t)j * DSML]);
  smaxp[b * DSML + c] = m;
}

// ---------- init class tokens ----------
__global__ void init_class(const float* __restrict__ xl, const float* __restrict__ xs,
                           float* __restrict__ clsL, float* __restrict__ clsS) {
  int idx = blockIdx.x * 256 + threadIdx.x;
  if (idx < NB * DLRG) {
    int b = idx >> 9, k = idx & 511;
    clsL[idx] = xl[(size_t)b * 2049 * DLRG + k];
  }
  int i2 = idx - NB * DLRG;
  if (i2 >= 0 && i2 < NB * DSML) {
    int b = i2 >> 10, k = i2 & 1023;
    clsS[i2] = xs[(size_t)b * 513 * DSML + k];
  }
}

// ---------- net_w fp32 -> bf16 ----------
__global__ void convert_netw(const float* __restrict__ nw, unsigned short* __restrict__ wnb) {
  int idx = blockIdx.x * 256 + threadIdx.x;   // packs 2 elements
  if (idx < 1024 * 512 / 2) {
    unsigned p = (unsigned)f2bf(nw[2 * idx]) | ((unsigned)f2bf(nw[2 * idx + 1]) << 16);
    ((unsigned*)wnb)[idx] = p;
  }
}

// ---------- matvec: y[b][o] = bias[o] + sum_i W[o][i]*x[b][i], all 32 batches/wave ----------
__global__ void matvec_all(const float* __restrict__ W, const float* __restrict__ bias,
                           const float* __restrict__ x, float* __restrict__ y,
                           int dout, int din) {
  int o = blockIdx.x * 8 + (threadIdx.x >> 5);
  int lane = threadIdx.x & 31;
  if (o >= dout) return;
  float acc[NB];
#pragma unroll
  for (int b = 0; b < NB; ++b) acc[b] = 0.f;
  const float* wr = W + (size_t)o * din;
  for (int i = lane; i < din; i += 32) {
    float w = wr[i];
#pragma unroll
    for (int b = 0; b < NB; ++b) acc[b] += w * x[b * din + i];
  }
  float mine = 0.f;
#pragma unroll
  for (int b = 0; b < NB; ++b) {
    float v = wave_sum(acc[b]);
    if (lane == b) mine = v;
  }
  if (bias) mine += bias[o];
  y[lane * dout + o] = mine;
}

// ---------- per-batch-vector layernorm ----------
__global__ void layernorm_vec(const float* __restrict__ x, const float* __restrict__ g,
                              const float* __restrict__ bt, float* __restrict__ y, int dim) {
  int b = blockIdx.x, t = threadIdx.x;
  const float* xb = x + (size_t)b * dim;
  __shared__ float s1[256], s2[256], mr[2];
  float sm = 0.f, sq = 0.f;
  for (int i = t; i < dim; i += 256) { float v = xb[i]; sm += v; sq += v * v; }
  s1[t] = sm; s2[t] = sq; __syncthreads();
  for (int s = 128; s; s >>= 1) {
    if (t < s) { s1[t] += s1[t + s]; s2[t] += s2[t + s]; }
    __syncthreads();
  }
  if (!t) {
    float m = s1[0] / dim;
    float var = s2[0] / dim - m * m;
    mr[0] = m; mr[1] = rsqrtf(var + 1e-5f);
  }
  __syncthreads();
  float m = mr[0], rs = mr[1];
  for (int i = t; i < dim; i += 256)
    y[(size_t)b * dim + i] = (xb[i] - m) * rs * g[i] + bt[i];
}

// ---------- tq/u: tq[b][h][j] = inv_sqrt_dk * sum_i Wk[h*dk+i][j]*q[b][h*dk+i]; u = tq*g ----------
__global__ void tqu_kernel(const float* __restrict__ Wk, const float* __restrict__ q,
                           const float* __restrict__ g, float* __restrict__ tq,
                           float* __restrict__ u, int d, int dk) {
  int j = blockIdx.x * 256 + threadIdx.x;
  int b = blockIdx.y;
  if (j >= d) return;
  float inv = rsqrtf((float)dk);
  for (int h = 0; h < 4; ++h) {
    float acc = 0.f;
    const float* qh = q + (size_t)b * d + h * dk;
    for (int i = 0; i < dk; ++i) acc += Wk[(size_t)(h * dk + i) * d + j] * qh[i];
    acc *= inv;
    tq[((size_t)b * 4 + h) * d + j] = acc;
    u [((size_t)b * 4 + h) * d + j] = acc * g[j];
  }
}

// ---------- scalar constants per (b,h): {c_h, s0_h} ----------
__global__ void attn_consts(const float* __restrict__ tq, const float* __restrict__ q,
                            const float* __restrict__ bk, const float* __restrict__ bca,
                            const float* __restrict__ hcls, float* __restrict__ cst,
                            int d, int dk) {
  int b = blockIdx.x;
  int h = threadIdx.x >> 5, lane = threadIdx.x & 31;   // 128 threads: 4 waves
  float a1 = 0.f, a2 = 0.f, a3 = 0.f;
  const float* tqh = tq + ((size_t)b * 4 + h) * d;
  for (int i = lane; i < d; i += 32) { float t = tqh[i]; a1 += t * bca[i]; a2 += t * hcls[(size_t)b * d + i]; }
  for (int i = lane; i < dk; i += 32) a3 += q[(size_t)b * d + h * dk + i] * bk[h * dk + i];
  a1 = wave_sum(a1); a2 = wave_sum(a2); a3 = wave_sum(a3);
  if (!lane) {
    float qb = a3 * rsqrtf((float)dk);
    cst[((size_t)b * 4 + h) * 2 + 0] = a1 + qb;   // patch score offset
    cst[((size_t)b * 4 + h) * 2 + 1] = a2 + qb;   // class token score
  }
}

// ---------- attention: scores, softmax, cross-head renorm, weighted sums ----------
__global__ void attn_kernel(const unsigned short* __restrict__ xhat, const float* __restrict__ u,
                            const float* __restrict__ cst, const float* __restrict__ hcls,
                            const float* __restrict__ gca, const float* __restrict__ bca,
                            float* __restrict__ y, float* __restrict__ sumA, int n, int d) {
  extern __shared__ float lds[];
  float* u_s = lds;              // 4*d
  float* sc  = lds + 4 * d;      // 4*n
  float* red = sc + 4 * n;       // 256
  float* msc = red + 256;        // 16
  int b = blockIdx.x, tid = threadIdx.x, lane = tid & 31, wid = tid >> 5;
  for (int i = tid; i < 4 * d; i += 256) u_s[i] = u[(size_t)b * 4 * d + i];
  __syncthreads();

  const unsigned short* xb = xhat + (size_t)b * (n - 1) * d;
  // ---- phase 1: patch scores (one wave per token) ----
  for (int j = wid; j < n - 1; j += 8) {
    const unsigned short* xr = xb + (size_t)j * d;
    float a0 = 0.f, a1 = 0.f, a2 = 0.f, a3 = 0.f;
    for (int i = lane; i < d; i += 32) {
      float x = bf2f(xr[i]);
      a0 += u_s[i] * x; a1 += u_s[d + i] * x; a2 += u_s[2 * d + i] * x; a3 += u_s[3 * d + i] * x;
    }
    a0 = wave_sum(a0); a1 = wave_sum(a1); a2 = wave_sum(a2); a3 = wave_sum(a3);
    if (!lane) {
      sc[0 * n + j + 1] = a0 + cst[((size_t)b * 4 + 0) * 2];
      sc[1 * n + j + 1] = a1 + cst[((size_t)b * 4 + 1) * 2];
      sc[2 * n + j + 1] = a2 + cst[((size_t)b * 4 + 2) * 2];
      sc[3 * n + j + 1] = a3 + cst[((size_t)b * 4 + 3) * 2];
    }
  }
  if (tid < 4) sc[tid * n] = cst[((size_t)b * 4 + tid) * 2 + 1];
  __syncthreads();

  // ---- per-head softmax ----
  for (int h = 0; h < 4; ++h) {
    float m = -3.402823466e38f;
    for (int j = tid; j < n; j += 256) m = fmaxf(m, sc[h * n + j]);
    red[tid] = m; __syncthreads();
    for (int s = 128; s; s >>= 1) { if (tid < s) red[tid] = fmaxf(red[tid], red[tid + s]); __syncthreads(); }
    float M = red[0]; __syncthreads();
    float sm = 0.f;
    for (int j = tid; j < n; j += 256) { float p = __expf(sc[h * n + j] - M); sc[h * n + j] = p; sm += p; }
    red[tid] = sm; __syncthreads();
    for (int s = 128; s; s >>= 1) { if (tid < s) red[tid] += red[tid + s]; __syncthreads(); }
    if (!tid) msc[4 + h] = red[0];
    __syncthreads();
  }

  // ---- cross-head renormalize; accumulate per-head patch attention mass ----
  float iS0 = 1.f / msc[4], iS1 = 1.f / msc[5], iS2 = 1.f / msc[6], iS3 = 1.f / msc[7];
  float pA[4] = {0.f, 0.f, 0.f, 0.f};
  for (int j = tid; j < n; j += 256) {
    float p0 = sc[j] * iS0, p1 = sc[n + j] * iS1, p2 = sc[2 * n + j] * iS2, p3 = sc[3 * n + j] * iS3;
    float r = 1.f / (1e-9f + p0 + p1 + p2 + p3);
    p0 *= r; p1 *= r; p2 *= r; p3 *= r;
    sc[j] = p0; sc[n + j] = p1; sc[2 * n + j] = p2; sc[3 * n + j] = p3;
    if (j > 0) { pA[0] += p0; pA[1] += p1; pA[2] += p2; pA[3] += p3; }
  }
  for (int h = 0; h < 4; ++h) {
    __syncthreads();
    red[tid] = pA[h]; __syncthreads();
    for (int s = 128; s; s >>= 1) { if (tid < s) red[tid] += red[tid + s]; __syncthreads(); }
    if (!tid) msc[8 + h] = red[0];
  }
  __syncthreads();

  // ---- phase 2: weighted sums over patch tokens (coalesced by column) ----
  int nslots = d >> 8;                     // 2 (d=512) or 4 (d=1024)
  float acc[4][4];
#pragma unroll
  for (int s = 0; s < 4; ++s)
#pragma unroll
    for (int h = 0; h < 4; ++h) acc[s][h] = 0.f;
  for (int j = 1; j < n; ++j) {
    float A0 = sc[j], A1 = sc[n + j], A2 = sc[2 * n + j], A3 = sc[3 * n + j];
    const unsigned short* xr = xb + (size_t)(j - 1) * d;
#pragma unroll
    for (int s = 0; s < 4; ++s) {
      if (s < nslots) {
        float x = bf2f(xr[tid + 256 * s]);
        acc[s][0] += A0 * x; acc[s][1] += A1 * x; acc[s][2] += A2 * x; acc[s][3] += A3 * x;
      }
    }
  }
  // ---- finalize y_h = A_h0*hcls + wsum_h*g + patchMass_h*b ----
#pragma unroll
  for (int s = 0; s < 4; ++s) {
    if (s < nslots) {
      int c = tid + 256 * s;
      float hc = hcls[(size_t)b * d + c], gg = gca[c], bb = bca[c];
#pragma unroll
      for (int h = 0; h < 4; ++h) {
        float A0h = sc[h * n];
        y[((size_t)b * 4 + h) * d + c] = A0h * hc + acc[s][h] * gg + msc[8 + h] * bb;
      }
    }
  }
  if (tid < 4) sumA[b * 4 + tid] = msc[8 + tid] + sc[tid * n];
}

// ---------- out[b][o] = calq[b][o] + sum_j Wv[o][j]*y[b][h][j] + sumA[b][h]*bv[o] ----------
__global__ void wv_apply(const float* __restrict__ Wv, const float* __restrict__ bv,
                         const float* __restrict__ y, const float* __restrict__ sumA,
                         const float* __restrict__ calq, float* __restrict__ outp,
                         int d, int dk) {
  int o = blockIdx.x * 8 + (threadIdx.x >> 5);
  int lane = threadIdx.x & 31;
  if (o >= d) return;
  int h = o / dk;
  float acc[NB];
#pragma unroll
  for (int b = 0; b < NB; ++b) acc[b] = 0.f;
  const float* wr = Wv + (size_t)o * d;
  for (int i = lane; i < d; i += 32) {
    float w = wr[i];
#pragma unroll
    for (int b = 0; b < NB; ++b) acc[b] += w * y[((size_t)b * 4 + h) * d + i];
  }
  float mine = 0.f;
#pragma unroll
  for (int b = 0; b < NB; ++b) {
    float v = wave_sum(acc[b]);
    if (lane == b) mine = v;
  }
  mine += calq[(size_t)lane * d + o] + sumA[lane * 4 + h] * bv[o];
  outp[(size_t)lane * d + o] = mine;
}

// ---------- WMMA bf16 GEMM, async double-buffered A tiles, fused bn_lrelu + col-max ----------
__global__ void gemm_max_kernel(const unsigned short* __restrict__ xlb,
                                const unsigned short* __restrict__ wb,
                                const float* __restrict__ bn_g, const float* __restrict__ bn_b,
                                float* __restrict__ lvec) {
  extern __shared__ char smem[];
  unsigned short* Bt = (unsigned short*)smem;     // [64][528] bf16
  unsigned short* At = Bt + 64 * 528;             // [2][32][528] bf16 (double buffer)
  float* cmax = (float*)(At + 2 * 32 * 528);      // [8][16]
  int b = blockIdx.y, colBase = blockIdx.x * 64;
  int tid = threadIdx.x, lane = tid & 31, wid = tid >> 5;
  int rg = wid >> 2, ct = wid & 3;

  // async-stage B tile (16 chunks/thread) and A tile 0 (8 chunks/thread)
  const u4_t* wsrc = (const u4_t*)(wb + (size_t)colBase * 512);
  for (int idx = tid; idx < 4096; idx += 256) {
    int r = idx >> 6, kc = idx & 63;
    async_g2l_b128(Bt + r * 528 + kc * 8, wsrc + idx);
  }
  const size_t xbase = (size_t)b * NLP * 512;
  {
    const u4_t* asrc = (const u4_t*)(xlb + xbase);
    for (int idx = tid; idx < 2048; idx += 256) {
      int r = idx >> 6, kc = idx & 63;
      async_g2l_b128(At + r * 528 + kc * 8, asrc + idx);
    }
  }

  int myCol = colBase + ct * 16 + (lane & 15);
  float gs = bn_g[myCol] * rsqrtf(1.f + 1e-5f);
  float bb = bn_b[myCol];
  float rm[8];
#pragma unroll
  for (int r = 0; r < 8; ++r) rm[r] = -3.402823466e38f;

  for (int iter = 0; iter < 64; ++iter) {
    unsigned short* Ac = At + (iter & 1) * (32 * 528);
    if (iter < 63) {
      // issue next tile while current batch finishes; buffer safe: last read
      // ended at previous iteration's trailing barrier.
      unsigned short* An = At + ((iter + 1) & 1) * (32 * 528);
      const u4_t* asrc = (const u4_t*)(xlb + xbase + (size_t)(iter + 1) * 32 * 512);
      for (int idx = tid; idx < 2048; idx += 256) {
        int r = idx >> 6, kc = idx & 63;
        async_g2l_b128(An + r * 528 + kc * 8, asrc + idx);
      }
      wait_async8();    // in-order: waits everything except the 8 just issued
    } else {
      wait_async0();
    }
    __syncthreads();

    v8f acc = {0.f, 0.f, 0.f, 0.f, 0.f, 0.f, 0.f, 0.f};
#pragma unroll
    for (int c = 0; c < 16; ++c) {
      int k0 = c * 32;
      AFrag af, bf;
      int ar = 16 * rg + (lane & 15);
      int ka = k0 + ((lane >> 4) << 3);                       // +0 / +8
      af.f[0] = *(const f4_t*)(Ac + ar * 528 + ka);
      af.f[1] = *(const f4_t*)(Ac + ar * 528 + ka + 16);
      int bc = ct * 16 + (lane & 15);
      int kb = k0 + ((lane >> 4) << 3);                       // +0 / +8
      bf.f[0] = *(const f4_t*)(Bt + bc * 528 + kb);
      bf.f[1] = *(const f4_t*)(Bt + bc * 528 + kb + 16);
      acc = __builtin_amdgcn_wmma_f32_16x16x32_bf16(false, af.v, false, bf.v,
                                                    (short)0, acc, false, false);
    }
#pragma unroll
    for (int r = 0; r < 8; ++r) {
      float yv = acc[r] * gs + bb;
      yv = yv >= 0.f ? yv : 0.2f * yv;
      rm[r] = fmaxf(rm[r], yv);
    }
    __syncthreads();   // all waves done reading Ac before it is overwritten
  }
  float m = rm[0];
#pragma unroll
  for (int r = 1; r < 8; ++r) m = fmaxf(m, rm[r]);
  m = fmaxf(m, __shfl_xor(m, 16, 32));       // lanes l and l^16 share N
  if (lane < 16) cmax[wid * 16 + lane] = m;
  __syncthreads();
  if (tid < 64) {
    int c = tid >> 4, nn = tid & 15;
    float v = fmaxf(cmax[c * 16 + nn], cmax[(c + 4) * 16 + nn]);
    lvec[b * 1024 + colBase + c * 16 + nn] = v;
  }
}

// ---------- merge class row into l; build fuse input f = l + s ----------
__global__ void merge_kernel(const float* __restrict__ ycls, const float* __restrict__ bn_g,
                             const float* __restrict__ bn_b, const float* __restrict__ smaxp,
                             const float* __restrict__ clsS, float* __restrict__ lvec,
                             float* __restrict__ fin) {
  int idx = blockIdx.x * 256 + threadIdx.x;   // 32*1024
  if (idx >= NB * 1024) return;
  int c = idx & 1023;
  float yv = ycls[idx] * (bn_g[c] * rsqrtf(1.f + 1e-5f)) + bn_b[c];
  yv = yv >= 0.f ? yv : 0.2f * yv;
  float l = fmaxf(lvec[idx], yv);
  float s = fmaxf(smaxp[idx], clsS[idx]);
  fin[idx] = l + s;
}

__global__ void fuse_out(const float* __restrict__ t2, const float* __restrict__ g,
                         const float* __restrict__ bb, float* __restrict__ out2) {
  int idx = blockIdx.x * 256 + threadIdx.x;   // 32*64
  if (idx >= NB * 64) return;
  int o = idx & 63;
  float y = t2[idx] * (g[o] * rsqrtf(1.f + 1e-5f)) + bb[o];
  out2[idx] = y >= 0.f ? y : 0.2f * y;
}

// =======================================================================
extern "C" void kernel_launch(void* const* d_in, const int* in_sizes, int n_in,
                              void* d_out, int out_size, void* d_ws, size_t ws_size,
                              hipStream_t stream) {
  (void)in_sizes; (void)n_in; (void)out_size; (void)ws_size;
  auto F = [&](int i) -> const float* { return (const float*)d_in[i]; };

  // leaf order: xl, xs, then sorted-key pytree flatten of params
  const float* xl = F(0);
  const float* xs = F(1);
  // per-enc: layers(4x24) then ln_ls1{b,g} ln_ls2{b,g} ln_sl1{b,g} ln_sl2{b,g}
  auto encBase = [](int e) { return 2 + e * 104; };
  auto layBase = [&](int e, int l) { return encBase(e) + l * 24; };
  const float* fuse_bn_b = F(210); const float* fuse_bn_g = F(211);
  const float* fuse_w = F(212);
  const float* net_bn_b = F(213); const float* net_bn_g = F(214);
  const float* net_w = F(215);

  float* out = (float*)d_out;
  float* out0 = out;                               // xl patches: 32*2048*512
  float* out1 = out + (size_t)NB * NLP * DLRG;     // xs patches: 32*512*1024
  float* out2 = out1 + (size_t)NB * NSP * DSML;    // ht: 32*64

  // workspace carve-up
  char* wsp = (char*)d_ws;
  auto alloc = [&](size_t bytes) -> void* {
    void* p = (void*)wsp;
    wsp += (bytes + 255) & ~(size_t)255;
    return p;
  };
  unsigned short* XLB = (unsigned short*)alloc((size_t)NB * NLP * DLRG * 2);
  unsigned short* XHL = (unsigned short*)alloc((size_t)NB * NLP * DLRG * 2);
  unsigned short* XHS = (unsigned short*)alloc((size_t)NB * NSP * DSML * 2);
  unsigned short* WNB = (unsigned short*)alloc((size_t)1024 * 512 * 2);
  float* CLS_L = (float*)alloc((size_t)NB * DLRG * 4);
  float* CLS_S = (float*)alloc((size_t)NB * DSML * 4);
  float* SMAXP = (float*)alloc((size_t)NB * 1024 * 4);
  float* LVEC  = (float*)alloc((size_t)NB * 1024 * 4);
  float* FIN   = (float*)alloc((size_t)NB * 1024 * 4);
  float* T0    = (float*)alloc((size_t)NB * 1024 * 4);
  float* TCALQ = (float*)alloc((size_t)NB * 1024 * 4);
  float* THCLS = (float*)alloc((size_t)NB * 1024 * 4);
  float* TQv   = (float*)alloc((size_t)NB * 1024 * 4);
  float* TTQ   = (float*)alloc((size_t)NB * 4 * 1024 * 4);
  float* TU    = (float*)alloc((size_t)NB * 4 * 1024 * 4);
  float* TCON  = (float*)alloc((size_t)NB * 4 * 2 * 4);
  float* TY    = (float*)alloc((size_t)NB * 4 * 1024 * 4);
  float* TSA   = (float*)alloc((size_t)NB * 4 * 4);
  float* TOUT  = (float*)alloc((size_t)NB * 1024 * 4);
  float* TOUT2 = (float*)alloc((size_t)NB * 1024 * 4);

  // ---- prep ----
  prep_large<<<NB * NLP, 128, 0, stream>>>(xl, out0, XLB, XHL);
  prep_small<<<NB * NSP, 256, 0, stream>>>(xs, out1, XHS);
  smax_kernel<<<dim3(4, NB), 256, 0, stream>>>(xs, SMAXP);
  init_class<<<(NB * DLRG + NB * DSML + 255) / 256, 256, 0, stream>>>(xl, xs, CLS_L, CLS_S);
  convert_netw<<<(1024 * 512 / 2 + 255) / 256, 256, 0, stream>>>(net_w, WNB);

  // ---- branch runner ----
  auto run_branch = [&](int caB, const float* c1w, const float* c1b, int d_att,
                        const float* ln1g, const float* ln1b,
                        const float* ln2g, const float* ln2b,
                        const float* c2w, const float* c2b, int d_cls,
                        float* cls, const unsigned short* xhat, int n) {
    int dk = d_att / 4;
    const float* kb = F(caB + 0); const float* kw = F(caB + 1);
    const float* lnb = F(caB + 2); const float* lng = F(caB + 3);
    const float* qb = F(caB + 4); const float* qw = F(caB + 5);
    const float* vb = F(caB + 6); const float* vw = F(caB + 7);

    matvec_all<<<d_att / 8, 256, 0, stream>>>(c1w, c1b, cls, T0, d_att, d_cls);
    layernorm_vec<<<NB, 256, 0, stream>>>(T0, ln1g, ln1b, TCALQ, d_att);
    layernorm_vec<<<NB, 256, 0, stream>>>(TCALQ, lng, lnb, THCLS, d_att);
    matvec_all<<<d_att / 8, 256, 0, stream>>>(qw, qb, THCLS, TQv, d_att, d_att);
    tqu_kernel<<<dim3(d_att / 256, NB), 256, 0, stream>>>(kw, TQv, lng, TTQ, TU, d_att, dk);
    attn_consts<<<NB, 128, 0, stream>>>(TTQ, TQv, kb, lnb, THCLS, TCON, d_att, dk);
    size_t albytes = (size_t)(4 * d_att + 4 * n + 256 + 16) * 4;
    attn_kernel<<<NB, 256, albytes, stream>>>(xhat, TU, TCON, THCLS, lng, lnb, TY, TSA, n, d_att);
    wv_apply<<<d_att / 8, 256, 0, stream>>>(vw, vb, TY, TSA, TCALQ, TOUT, d_att, dk);
    matvec_all<<<d_cls / 8, 256, 0, stream>>>(c2w, c2b, TOUT, TOUT2, d_cls, d_att);
    layernorm_vec<<<NB, 256, 0, stream>>>(TOUT2, ln2g, ln2b, cls, d_cls);
  };

  // ---- 2 encoders x 4 layers ----
  for (int e = 0; e < 2; ++e) {
    int eb = encBase(e);
    const float* ln_ls1_b = F(eb + 96); const float* ln_ls1_g = F(eb + 97);
    const float* ln_ls2_b = F(eb + 98); const float* ln_ls2_g = F(eb + 99);
    const float* ln_sl1_b = F(eb + 100); const float* ln_sl1_g = F(eb + 101);
    const float* ln_sl2_b = F(eb + 102); const float* ln_sl2_g = F(eb + 103);
    for (int l = 0; l < 4; ++l) {
      int lb = layBase(e, l);
      // branch A: large class attends to small patches (d_att=1024, ca_l)
      run_branch(lb + 0, F(lb + 17), F(lb + 16), DSML,
                 ln_ls1_g, ln_ls1_b, ln_sl1_g, ln_sl1_b,
                 F(lb + 23), F(lb + 22), DLRG, CLS_L, XHS, NSP + 1);
      // branch B: small class attends to large patches (d_att=512, ca_s)
      run_branch(lb + 8, F(lb + 19), F(lb + 18), DLRG,
                 ln_sl2_g, ln_sl2_b, ln_ls2_g, ln_ls2_b,
                 F(lb + 21), F(lb + 20), DSML, CLS_S, XHL, NLP + 1);
    }
  }

  // ---- final head ----
  size_t gemm_lds = (size_t)(64 * 528 + 2 * 32 * 528) * 2 + 8 * 16 * 4;
  gemm_max_kernel<<<dim3(16, NB), 256, gemm_lds, stream>>>(XLB, WNB, net_bn_g, net_bn_b, LVEC);
  matvec_all<<<1024 / 8, 256, 0, stream>>>(net_w, nullptr, CLS_L, T0, 1024, 512);
  merge_kernel<<<(NB * 1024 + 255) / 256, 256, 0, stream>>>(T0, net_bn_g, net_bn_b, SMAXP, CLS_S, LVEC, FIN);
  matvec_all<<<64 / 8, 256, 0, stream>>>(fuse_w, nullptr, FIN, TQv, 64, 1024);
  fuse_out<<<(NB * 64 + 255) / 256, 256, 0, stream>>>(TQv, fuse_bn_g, fuse_bn_b, out2);
}
